// GAT_66872640799457
// MI455X (gfx1250) — compile-verified, matrix operand order checked by reading, not run on hardware
//
#include <hip/hip_runtime.h>
#include <hip/hip_bf16.h>
#include <stdint.h>

#define NNODES   100000
#define FDIM     128
#define NCLS     40
#define NEG_SLOPE 0.2f
#define BN_EPS   1e-5f

typedef __bf16 bf16_t;
typedef bf16_t v16bf __attribute__((ext_vector_type(16)));
typedef float  v8f   __attribute__((ext_vector_type(8)));

union Frag32B { uint4 u4[2]; v16bf v; };

// gfx1250 async global->LDS path (ASYNCcnt-tracked), guarded so compile is
// safe if this toolchain doesn't declare the builtins.
#if defined(__gfx1250__) && \
    __has_builtin(__builtin_amdgcn_global_load_async_to_lds_b128) && \
    __has_builtin(__builtin_amdgcn_s_wait_asynccnt)
#define HAVE_ASYNC_LDS 1
#else
#define HAVE_ASYNC_LDS 0
#endif

// exact param types per hipcc diagnostic: GCC vector int4, AS1 src / AS3 dst
typedef int v4i_b128 __attribute__((vector_size(16)));
typedef __attribute__((address_space(1))) v4i_b128 gv4i_t;
typedef __attribute__((address_space(3))) v4i_b128 lv4i_t;

// ---------- helpers ----------
static __device__ __forceinline__ unsigned short f32_to_bf16(float f) {
    unsigned int u = __float_as_uint(f);
    unsigned int r = (u + 0x7FFFu + ((u >> 16) & 1u)) >> 16;   // RNE
    return (unsigned short)r;
}

static __device__ __forceinline__ float lrelu(float a) {
    return a > 0.0f ? a : NEG_SLOPE * a;
}

static __device__ __forceinline__ void atomicMaxFloat(float* addr, float val) {
    // positive floats order as signed ints; negative floats order inversely
    // as unsigned ints. Seeded with a finite float, mixed updates stay correct.
    if (val >= 0.0f) atomicMax((int*)addr, __float_as_int(val));
    else             atomicMin((unsigned int*)addr, __float_as_uint(val));
}

// ---------- conversion / packing ----------
__global__ void k_f32_to_bf16(const float* __restrict__ in,
                              unsigned short* __restrict__ out, size_t n) {
    size_t i = (size_t)blockIdx.x * blockDim.x + threadIdx.x;
    if (i < n) out[i] = f32_to_bf16(in[i]);
}

// Repack weight W (row-major K=128 x Ncols) into WMMA B-fragment order:
// fragment (kt,nt) = 32 lanes x 16 bf16 contiguous. Lane = column within
// 16-col tile; lane hi-half holds K offsets 8..15 / 24..31 per the
// gfx1250 16-bit operand VGPR layout. Zero-pad past Ncols.
__global__ void k_pack_w(const float* __restrict__ W,
                         unsigned short* __restrict__ Bp, int Ncols, int NT) {
    int tid = blockIdx.x * blockDim.x + threadIdx.x;
    int total = 4 * NT * 32 * 16;
    if (tid >= total) return;
    int e    = tid & 15;
    int lane = (tid >> 4) & 31;
    int fi   = tid >> 9;
    int nt   = fi % NT;
    int kt   = fi / NT;
    int l  = lane & 15, hi = lane >> 4;
    int koff = (e < 8 ? e : e + 8) + (hi ? 8 : 0);
    int kg = kt * 32 + koff;
    int ng = nt * 16 + l;
    float w = (ng < Ncols) ? W[kg * Ncols + ng] : 0.0f;
    Bp[tid] = f32_to_bf16(w);
}

// ---------- WMMA GEMM: C[nrows x ncols] = A[nrows x 128] * Bp (+bias) ----------
// Packed B (4*NT fragments, 32KB max) is staged once per block into LDS via
// async global->LDS copies; waves then read fragments with ds_load_b128.
template <int NT>
__global__ __launch_bounds__(256)
void k_wmma_gemm(const unsigned short* __restrict__ A,
                 const unsigned short* __restrict__ Bp,
                 float* __restrict__ C, const float* __restrict__ bias,
                 int nrows, int ldc, int ncols) {
    __shared__ __align__(16) unsigned short sB[4 * NT * 32 * 16];
    constexpr int CHUNKS = (4 * NT * 32 * 16) / 8;   // 16-byte chunks

    for (int i = threadIdx.x; i < CHUNKS; i += (int)blockDim.x) {
#if HAVE_ASYNC_LDS
        __builtin_amdgcn_global_load_async_to_lds_b128(
            (gv4i_t*)(Bp + (size_t)i * 8),
            (lv4i_t*)(sB + (size_t)i * 8), 0, 0);
#else
        ((uint4*)sB)[i] = ((const uint4*)Bp)[i];
#endif
    }
#if HAVE_ASYNC_LDS
    __builtin_amdgcn_s_wait_asynccnt(0);
#endif
    __syncthreads();

    const int wave = threadIdx.x >> 5;
    const int lane = threadIdx.x & 31;
    const int row0 = (blockIdx.x * (blockDim.x >> 5) + wave) * 16;
    if (row0 >= nrows) return;                 // wave-uniform: EXEC stays all-1s
    const int l  = lane & 15;
    const int hi = lane >> 4;

    // A fragments for full K=128: lane holds row (row0+l), K chunks per hi half
    Frag32B a[4];
    const size_t rbase = (size_t)(row0 + l) * FDIM;
#pragma unroll
    for (int kt = 0; kt < 4; ++kt) {
        const uint4* p = (const uint4*)(A + rbase + kt * 32 + hi * 8);
        a[kt].u4[0] = p[0];      // K local 0..7   (this half)
        a[kt].u4[1] = p[2];      // K local 16..23 (this half) : +32 bytes
    }

#pragma unroll
    for (int nt = 0; nt < NT; ++nt) {
        v8f c = {0.f, 0.f, 0.f, 0.f, 0.f, 0.f, 0.f, 0.f};
#pragma unroll
        for (int kt = 0; kt < 4; ++kt) {
            Frag32B b;
            const uint4* bp =
                (const uint4*)(sB + ((size_t)(kt * NT + nt) * 32 + lane) * 16);
            b.u4[0] = bp[0];
            b.u4[1] = bp[1];
            c = __builtin_amdgcn_wmma_f32_16x16x32_bf16(
                    false, a[kt].v, false, b.v, (short)0, c, false, false);
        }
        const int colb = nt * 16 + l;
        if (colb < ncols) {
            float badd = bias ? bias[colb] : 0.0f;
#pragma unroll
            for (int j = 0; j < 8; ++j) {
                // C/D layout: vgpr j, lanes 0-15 -> M=j ; lanes 16-31 -> M=j+8
                C[(size_t)(row0 + j + 8 * hi) * ldc + colb] = c[j] + badd;
            }
        }
    }
}

// ---------- attention score prep: wave per node ----------
__global__ void k_alpha(const float* __restrict__ h,
                        const float* __restrict__ avs, const float* __restrict__ avd,
                        float* __restrict__ alpha_s, float* __restrict__ alpha_d,
                        float* __restrict__ m, float* __restrict__ s, int N) {
    int node = (blockIdx.x * blockDim.x + threadIdx.x) >> 5;
    int lane = threadIdx.x & 31;
    if (node >= N) return;
    const float4* hp = (const float4*)(h + (size_t)node * FDIM);
    const float4* ap = (const float4*)avs;
    const float4* dp = (const float4*)avd;
    float4 hv = hp[lane], av = ap[lane], dv = dp[lane];
    float ss = hv.x * av.x + hv.y * av.y + hv.z * av.z + hv.w * av.w;
    float sd = hv.x * dv.x + hv.y * dv.y + hv.z * dv.z + hv.w * dv.w;
#pragma unroll
    for (int o = 16; o > 0; o >>= 1) {
        ss += __shfl_xor(ss, o);
        sd += __shfl_xor(sd, o);
    }
    if (lane == 0) {
        alpha_s[node] = ss;
        alpha_d[node] = sd;
        m[node] = lrelu(ss + sd);   // self-loop score seeds the segment max
        s[node] = 0.0f;
    }
}

// ---------- edge passes ----------
__global__ void k_edge_max(const int* __restrict__ src, const int* __restrict__ dst,
                           const float* __restrict__ as, const float* __restrict__ ad,
                           float* __restrict__ m, int E) {
    int e = blockIdx.x * blockDim.x + threadIdx.x;
    if (e >= E) return;
    int d = dst[e];
    float a = lrelu(as[src[e]] + ad[d]);
    atomicMaxFloat(&m[d], a);
}

__global__ void k_edge_exp(const int* __restrict__ src, const int* __restrict__ dst,
                           const float* __restrict__ as, const float* __restrict__ ad,
                           const float* __restrict__ m, float* __restrict__ s,
                           float* __restrict__ ebuf, int E) {
    int e = blockIdx.x * blockDim.x + threadIdx.x;
    if (e >= E) return;
    int d = dst[e];
    float a = lrelu(as[src[e]] + ad[d]);
    float ev = __expf(a - m[d]);
    ebuf[e] = ev;
    atomicAdd(&s[d], ev);
}

// self-loop: finalize denominator, store normalized self weight
__global__ void k_self(const float* __restrict__ as, const float* __restrict__ ad,
                       const float* __restrict__ m, float* __restrict__ s,
                       float* __restrict__ coef, int N) {
    int i = blockIdx.x * blockDim.x + threadIdx.x;
    if (i >= N) return;
    float w  = __expf(lrelu(as[i] + ad[i]) - m[i]);
    float sn = s[i] + w;
    s[i]     = sn;
    coef[i]  = w / sn;
}

__global__ void k_self_agg(const float* __restrict__ h, const float* __restrict__ coef,
                           float* __restrict__ agg, size_t total) {
    size_t i = (size_t)blockIdx.x * blockDim.x + threadIdx.x;
    if (i < total) agg[i] = h[i] * coef[i >> 7];
}

// wave per edge, float4 per lane (32 lanes x 4 = 128 channels)
__global__ void k_edge_agg(const int* __restrict__ src, const int* __restrict__ dst,
                           const float* __restrict__ ebuf, const float* __restrict__ s,
                           const float* __restrict__ h, float* __restrict__ agg, int E) {
    int e = (blockIdx.x * blockDim.x + threadIdx.x) >> 5;
    int lane = threadIdx.x & 31;
    if (e >= E) return;
    int si = src[e], di = dst[e];
    float w = ebuf[e] / s[di];
    float4 v = ((const float4*)(h + (size_t)si * FDIM))[lane];
    float* ad = agg + (size_t)di * FDIM + lane * 4;
    atomicAdd(ad + 0, v.x * w);
    atomicAdd(ad + 1, v.y * w);
    atomicAdd(ad + 2, v.z * w);
    atomicAdd(ad + 3, v.w * w);
}

// ---------- batchnorm ----------
__global__ void k_bn_stats(const float* __restrict__ agg,
                           float* __restrict__ sums, float* __restrict__ sqs, int N) {
    int c = threadIdx.x & 127;
    int rl = (blockIdx.x * blockDim.x + threadIdx.x) >> 7;
    int nr = (gridDim.x * blockDim.x) >> 7;
    float s = 0.0f, q = 0.0f;
    for (int r = rl; r < N; r += nr) {
        float v = agg[(size_t)r * FDIM + c];
        s += v; q += v * v;
    }
    atomicAdd(&sums[c], s);
    atomicAdd(&sqs[c], q);
}

__global__ void k_bn_final(const float* __restrict__ sums, const float* __restrict__ sqs,
                           const float* __restrict__ g, const float* __restrict__ beta,
                           float* __restrict__ sc, float* __restrict__ sh, int N) {
    int c = threadIdx.x;
    if (c >= FDIM) return;
    float mu  = sums[c] / (float)N;                 // GAT bias cancels in BN
    float var = sqs[c] / (float)N - mu * mu;
    float a   = g[c] * rsqrtf(var + BN_EPS);
    sc[c] = a;
    sh[c] = beta[c] - mu * a;
}

__global__ void k_bn_relu_bf16(const float* __restrict__ agg,
                               const float* __restrict__ sc, const float* __restrict__ sh,
                               unsigned short* __restrict__ out, size_t total) {
    size_t i = (size_t)blockIdx.x * blockDim.x + threadIdx.x;
    if (i >= total) return;
    int c = (int)(i & 127);
    float v = fmaxf(agg[i] * sc[c] + sh[c], 0.0f);
    out[i] = f32_to_bf16(v);
}

// ---------- launch ----------
extern "C" void kernel_launch(void* const* d_in, const int* in_sizes, int n_in,
                              void* d_out, int out_size, void* d_ws, size_t ws_size,
                              hipStream_t stream) {
    const float* x    = (const float*)d_in[0];
    const int*   ei   = (const int*)d_in[1];
    const float* W1   = (const float*)d_in[2];
    const float* as1  = (const float*)d_in[3];
    const float* ad1  = (const float*)d_in[4];
    const float* g1   = (const float*)d_in[6];
    const float* be1  = (const float*)d_in[7];
    const float* W2   = (const float*)d_in[8];
    const float* as2  = (const float*)d_in[9];
    const float* ad2  = (const float*)d_in[10];
    const float* g2   = (const float*)d_in[12];
    const float* be2  = (const float*)d_in[13];
    const float* Wc   = (const float*)d_in[14];
    const float* bc   = (const float*)d_in[15];

    const int N = NNODES;
    const int E = in_sizes[1] / 2;
    const int* srcp = ei;
    const int* dstp = ei + E;

    // workspace carve-up (256B aligned)
    char* base = (char*)d_ws;
    size_t off = 0;
    auto carve = [&](size_t bytes) -> char* {
        char* p = base + off;
        off = (off + bytes + 255) & ~(size_t)255;
        return p;
    };
    float*          hbuf  = (float*)carve((size_t)N * FDIM * 4);
    float*          agg   = (float*)carve((size_t)N * FDIM * 4);
    unsigned short* bfb   = (unsigned short*)carve((size_t)N * FDIM * 2);
    float*          alS   = (float*)carve((size_t)N * 4);
    float*          alD   = (float*)carve((size_t)N * 4);
    float*          mbuf  = (float*)carve((size_t)N * 4);
    float*          sbuf  = (float*)carve((size_t)N * 4);
    float*          coef  = (float*)carve((size_t)N * 4);
    float*          ebuf  = (float*)carve((size_t)E * 4);
    unsigned short* Bp1   = (unsigned short*)carve(4 * 8 * 32 * 16 * 2);
    unsigned short* Bp2   = (unsigned short*)carve(4 * 8 * 32 * 16 * 2);
    unsigned short* Bpc   = (unsigned short*)carve(4 * 3 * 32 * 16 * 2);
    float*          sums  = (float*)carve(FDIM * 4);
    float*          sqs   = (float*)carve(FDIM * 4);
    float*          scv   = (float*)carve(FDIM * 4);
    float*          shv   = (float*)carve(FDIM * 4);

    const size_t tot = (size_t)N * FDIM;
    const int TB = 256;
    const int elB  = (int)((tot + TB - 1) / TB);
    const int edB  = (E + TB - 1) / TB;
    const int ndB  = (N + TB - 1) / TB;
    const int wvNd = (N * 32 + TB - 1) / TB;       // wave-per-node grids
    const int wvEd = (int)(((size_t)E * 32 + TB - 1) / TB);
    const int gmB  = ((N / 16) + 7) / 8;           // 8 waves (16-row tiles) / block

    // activations -> bf16, weights -> fragment layout
    k_f32_to_bf16<<<elB, TB, 0, stream>>>(x, bfb, tot);
    k_pack_w<<<(4 * 8 * 32 * 16 + TB - 1) / TB, TB, 0, stream>>>(W1, Bp1, FDIM, 8);
    k_pack_w<<<(4 * 8 * 32 * 16 + TB - 1) / TB, TB, 0, stream>>>(W2, Bp2, FDIM, 8);
    k_pack_w<<<(4 * 3 * 32 * 16 + TB - 1) / TB, TB, 0, stream>>>(Wc, Bpc, NCLS, 3);

    auto layer = [&](const float* aS, const float* aD, const float* g,
                     const float* beta, const unsigned short* Bp) {
        k_wmma_gemm<8><<<gmB, TB, 0, stream>>>(bfb, Bp, hbuf, nullptr, N, FDIM, FDIM);
        k_alpha<<<wvNd, TB, 0, stream>>>(hbuf, aS, aD, alS, alD, mbuf, sbuf, N);
        k_edge_max<<<edB, TB, 0, stream>>>(srcp, dstp, alS, alD, mbuf, E);
        k_edge_exp<<<edB, TB, 0, stream>>>(srcp, dstp, alS, alD, mbuf, sbuf, ebuf, E);
        k_self<<<ndB, TB, 0, stream>>>(alS, alD, mbuf, sbuf, coef, N);
        k_self_agg<<<elB, TB, 0, stream>>>(hbuf, coef, agg, tot);
        k_edge_agg<<<wvEd, TB, 0, stream>>>(srcp, dstp, ebuf, sbuf, hbuf, agg, E);
        (void)hipMemsetAsync(sums, 0, FDIM * sizeof(float), stream);
        (void)hipMemsetAsync(sqs, 0, FDIM * sizeof(float), stream);
        k_bn_stats<<<512, TB, 0, stream>>>(agg, sums, sqs, N);
        k_bn_final<<<1, FDIM, 0, stream>>>(sums, sqs, g, beta, scv, shv, N);
        k_bn_relu_bf16<<<elB, TB, 0, stream>>>(agg, scv, shv, bfb, tot);
    };

    layer(as1, ad1, g1, be1, Bp1);
    layer(as2, ad2, g2, be2, Bp2);

    // classifier: out = h2 @ Wc + bc  (40 cols, padded to 3 n-tiles)
    k_wmma_gemm<3><<<gmB, TB, 0, stream>>>(bfb, Bpc, (float*)d_out, bc, N, NCLS, NCLS);
    (void)n_in; (void)out_size; (void)ws_size;
}
